// S4DBottleneck_48369921688075
// MI455X (gfx1250) — compile-verified, matrix operand order checked by reading, not run on hardware
//
#include <hip/hip_runtime.h>
#include <hip/hip_bf16.h>

typedef __bf16 bf16_t;
typedef __attribute__((ext_vector_type(16))) __bf16 v16bf;
typedef __attribute__((ext_vector_type(4)))  __bf16 v4bf;
typedef __attribute__((ext_vector_type(8)))  float  v8f;

#define B_DIM 4
#define T_DIM 4096
#define F_DIM 256
#define CF    512      // input feature size (C*F)
#define H_DIM 1024     // hidden size
#define M_DIM (B_DIM*T_DIM)   // 16384 rows
#define NC    64       // scan chunks
#define TC    64       // chunk length (NC*TC == T_DIM)

// ---------------------------------------------------------------------------
// Async global -> LDS copy (CDNA5 GLOBAL_LOAD_ASYNC_TO_LDS, ASYNCcnt-tracked).
// INST_OFFSET is added to BOTH the LDS and the global address (ISA 15.18.3),
// so one address pair + offset:16 moves this thread's 32-byte slot.
// ---------------------------------------------------------------------------
__device__ __forceinline__ void async_b128_x2(unsigned lds_off, const void* gptr) {
  unsigned long long ga = (unsigned long long)(uintptr_t)gptr;
  asm volatile("global_load_async_to_lds_b128 %0, %1, off"
               :: "v"(lds_off), "v"(ga) : "memory");
  asm volatile("global_load_async_to_lds_b128 %0, %1, off offset:16"
               :: "v"(lds_off), "v"(ga) : "memory");
}

// ---------------------------------------------------------------------------
// f32 -> bf16 conversion kernels
// ---------------------------------------------------------------------------
__global__ void conv_u_kernel(const float* __restrict__ x, bf16_t* __restrict__ u) {
  int idx = blockIdx.x * blockDim.x + threadIdx.x;   // one thread per 4 elems
  long i4  = (long)idx << 2;
  int row = (int)(i4 >> 9);       // / 512
  int n   = (int)(i4 & 511);
  int b = row >> 12, t = row & 4095;
  int c = n >> 8,   f = n & 255;
  long src = (((long)(b * 2 + c) * T_DIM + t) * F_DIM + f);
  float4 xv = *reinterpret_cast<const float4*>(x + src);
  v4bf o = { (__bf16)xv.x, (__bf16)xv.y, (__bf16)xv.z, (__bf16)xv.w };
  *reinterpret_cast<v4bf*>(u + (long)row * CF + n) = o;
}

__global__ void conv_f32_bf16(const float* __restrict__ in, bf16_t* __restrict__ out, int n4) {
  int idx = blockIdx.x * blockDim.x + threadIdx.x;
  if (idx >= n4) return;
  float4 xv = *reinterpret_cast<const float4*>(in + ((long)idx << 2));
  v4bf o = { (__bf16)xv.x, (__bf16)xv.y, (__bf16)xv.z, (__bf16)xv.w };
  *reinterpret_cast<v4bf*>(out + ((long)idx << 2)) = o;
}

// ---------------------------------------------------------------------------
// One K-step of the wave-level WMMA compute from an LDS buffer pair.
// ---------------------------------------------------------------------------
__device__ __forceinline__ void wmma_step(const bf16_t* __restrict__ Ab,
                                          const bf16_t* __restrict__ Bb,
                                          v8f (&acc)[4][2],
                                          int wm, int wn, int mrow, int ksel) {
  v16bf afr[4], bfr[2];
#pragma unroll
  for (int mt = 0; mt < 4; ++mt) {
    const bf16_t* p = &Ab[(wm + mt * 16 + mrow) * 40 + ksel];
    uint4* d = reinterpret_cast<uint4*>(&afr[mt]);
    d[0] = *reinterpret_cast<const uint4*>(p);        // K sel .. sel+7
    d[1] = *reinterpret_cast<const uint4*>(p + 16);   // K sel+16 .. sel+23
  }
#pragma unroll
  for (int nt = 0; nt < 2; ++nt) {
    const bf16_t* p = &Bb[(wn + nt * 16 + mrow) * 40 + ksel];
    uint4* d = reinterpret_cast<uint4*>(&bfr[nt]);
    d[0] = *reinterpret_cast<const uint4*>(p);
    d[1] = *reinterpret_cast<const uint4*>(p + 16);
  }
#pragma unroll
  for (int mt = 0; mt < 4; ++mt)
#pragma unroll
    for (int nt = 0; nt < 2; ++nt)
      acc[mt][nt] = __builtin_amdgcn_wmma_f32_16x16x32_bf16(
          false, afr[mt], false, bfr[nt], (short)0, acc[mt][nt], false, false);
}

// ---------------------------------------------------------------------------
// WMMA GEMM: out = A(MxK, bf16, row-major) * W(NxK, bf16, row-major)^T
// block tile 128x128, 256 threads = 8 waves (2 M x 4 N), wave tile 64x32.
// Double-buffered LDS filled by async copies. The steady-state loop is kept
// ROLLED (#pragma unroll 1) and processes tiles in pairs so the buffer parity
// is a compile-time constant without full unrolling (keeps VGPR count low so
// more waves fit per SIMD). Invariant: <=8 async in flight per wave;
// s_wait_asynccnt 0x4 retires exactly the tile about to be computed.
// MODE 0: out[row*NSIZE+col] = acc + bias[col]                 (v = u@Win^T+b)
// MODE 1: d_out[perm] = acc + bias[col] + D[col]*x[perm]       (final output)
// ---------------------------------------------------------------------------
template <int K, int NSIZE, int MODE>
__global__ __launch_bounds__(256) void gemm_wmma(
    const bf16_t* __restrict__ A,
    const bf16_t* __restrict__ Bw,
    const float*  __restrict__ bias,
    float* __restrict__ out,
    const float* __restrict__ x,
    const float* __restrict__ Dvec)
{
  __shared__ __align__(16) bf16_t As[2][128 * 40];   // 32 cols + 8 pad
  __shared__ __align__(16) bf16_t Bs[2][128 * 40];

  const int tid  = threadIdx.x;
  const int lane = tid & 31;
  const int wave = tid >> 5;
  const int wm = (wave & 1) * 64;      // wave M offset in block tile
  const int wn = (wave >> 1) * 32;     // wave N offset in block tile
  const long rowBase = (long)blockIdx.x * 128;
  const long colBase = (long)blockIdx.y * 128;

  const int lr   = tid >> 1;           // 0..127 : tile row to stage
  const int lseg = (tid & 1) * 16;     // bf16 column offset 0 / 16

  // staging addresses (LDS byte offsets for the async engine)
  const unsigned ldsA0 = (unsigned)(uintptr_t)&As[0][lr * 40 + lseg];
  const unsigned ldsA1 = (unsigned)(uintptr_t)&As[1][lr * 40 + lseg];
  const unsigned ldsB0 = (unsigned)(uintptr_t)&Bs[0][lr * 40 + lseg];
  const unsigned ldsB1 = (unsigned)(uintptr_t)&Bs[1][lr * 40 + lseg];
  const bf16_t* gA = A  + (rowBase + lr) * (long)K + lseg;
  const bf16_t* gB = Bw + (colBase + lr) * (long)K + lseg;

  const v8f vzero = {0.f,0.f,0.f,0.f,0.f,0.f,0.f,0.f};
  v8f acc[4][2];
#pragma unroll
  for (int i = 0; i < 4; ++i)
#pragma unroll
    for (int j = 0; j < 2; ++j) acc[i][j] = vzero;

  // WMMA 16-bit A-matrix lane mapping (16x32):
  //   lanes 0-15 : M=lane,   elems 0..7 -> K 0..7,   elems 8..15 -> K 16..23
  //   lanes16-31 : M=lane-16,elems 0..7 -> K 8..15,  elems 8..15 -> K 24..31
  const int mrow = lane & 15;
  const int ksel = (lane >> 4) << 3;   // 0 or 8

  constexpr int nK = K >> 5;           // 16 or 32 (even, >= 4)

  // prologue: tile 0 -> buffer 0
  async_b128_x2(ldsA0, gA);
  async_b128_x2(ldsB0, gB);

  // steady state (rolled): pairs (kt, kt+1); last pair peeled below
#pragma unroll 1
  for (int kt = 0; kt < nK - 2; kt += 2) {
    const bf16_t* a1 = gA + ((kt + 1) << 5);
    const bf16_t* b1 = gB + ((kt + 1) << 5);
    async_b128_x2(ldsA1, a1);                         // tile kt+1 -> buf1
    async_b128_x2(ldsB1, b1);
    __builtin_prefetch(a1 + 64, 0, 1);                // warm L2 two tiles out
    __builtin_prefetch(b1 + 64, 0, 1);
    asm volatile("s_wait_asynccnt 0x4" ::: "memory"); // tile kt landed
    __syncthreads();
    wmma_step(As[0], Bs[0], acc, wm, wn, mrow, ksel);
    __syncthreads();                                  // buf0 free for refill

    async_b128_x2(ldsA0, gA + ((kt + 2) << 5));       // tile kt+2 -> buf0
    async_b128_x2(ldsB0, gB + ((kt + 2) << 5));
    asm volatile("s_wait_asynccnt 0x4" ::: "memory"); // tile kt+1 landed
    __syncthreads();
    wmma_step(As[1], Bs[1], acc, wm, wn, mrow, ksel);
    __syncthreads();                                  // buf1 free for refill
  }

  // tail pair: tile nK-2 already in flight to buf0
  async_b128_x2(ldsA1, gA + ((nK - 1) << 5));         // tile nK-1 -> buf1
  async_b128_x2(ldsB1, gB + ((nK - 1) << 5));
  asm volatile("s_wait_asynccnt 0x4" ::: "memory");   // tile nK-2 landed
  __syncthreads();
  wmma_step(As[0], Bs[0], acc, wm, wn, mrow, ksel);
  asm volatile("s_wait_asynccnt 0x0" ::: "memory");   // tile nK-1 landed
  __syncthreads();
  wmma_step(As[1], Bs[1], acc, wm, wn, mrow, ksel);

  // output: C/D layout -> VGPR j: lanes 0-15 = row j, lanes 16-31 = row j+8
  const int lhalf = (lane >> 4) << 3;
  const int lcol  = lane & 15;
#pragma unroll
  for (int mt = 0; mt < 4; ++mt) {
#pragma unroll
    for (int nt = 0; nt < 2; ++nt) {
      const long col = colBase + wn + nt * 16 + lcol;
      const float bv = bias[col];
#pragma unroll
      for (int j = 0; j < 8; ++j) {
        const long row = rowBase + wm + mt * 16 + lhalf + j;
        float val = acc[mt][nt][j] + bv;
        if (MODE == 0) {
          out[row * (long)NSIZE + col] = val;
        } else {
          int b = (int)(row >> 12), t = (int)(row & 4095);
          int cc = (int)(col >> 8), f = (int)(col & 255);
          long g = (((long)(b * 2 + cc) * T_DIM + t) * F_DIM + f);
          out[g] = val + Dvec[col] * x[g];
        }
      }
    }
  }
}

// ---------------------------------------------------------------------------
// S4D diagonal complex scan, chunk-parallel (3 passes)
// ---------------------------------------------------------------------------
__device__ __forceinline__ void s4d_coef(const float* Alog, const float* Ath,
                                         int h, float& ar, float& ai) {
  float sp = log1pf(expf(Alog[h]));
  sp = fmaxf(sp, 0.01f);
  float r = expf(-sp);
  float th = Ath[h];
  ar = r * cosf(th);
  ai = r * sinf(th);
}

// pass 1: per-chunk local scan with h0 = 0 -> S[b][c][h]
__global__ void scan_local(const float* __restrict__ v,
                           const float* __restrict__ Alog, const float* __restrict__ Ath,
                           const float* __restrict__ Brv,  const float* __restrict__ Biv,
                           float* __restrict__ Sr, float* __restrict__ Si) {
  int idx = blockIdx.x * 256 + threadIdx.x;   // (b*NC+c)*H + h
  int h  = idx & (H_DIM - 1);
  int bc = idx >> 10;
  int b = bc >> 6, c = bc & (NC - 1);
  float ar, ai; s4d_coef(Alog, Ath, h, ar, ai);
  float br = Brv[h], bi = Biv[h];
  long base = ((long)b * T_DIM + c * TC) * H_DIM + h;
  float hr = 0.f, hi = 0.f;
  for (int tl = 0; tl < TC; ++tl) {
    float vv = v[base + (long)tl * H_DIM];
    float nr = ar * hr - ai * hi + br * vv;
    float ni = ar * hi + ai * hr + bi * vv;
    hr = nr; hi = ni;
  }
  Sr[idx] = hr; Si[idx] = hi;
}

// pass 2: sequential combine over the 64 chunks: hstart[c+1] = a^TC*hstart[c]+S[c]
__global__ void scan_combine(const float* __restrict__ Sr, const float* __restrict__ Si,
                             const float* __restrict__ Alog, const float* __restrict__ Ath,
                             float* __restrict__ Hr, float* __restrict__ Hi) {
  int idx = blockIdx.x * 256 + threadIdx.x;   // b*H + h  (4096 total)
  int h = idx & (H_DIM - 1);
  int b = idx >> 10;
  float ar, ai; s4d_coef(Alog, Ath, h, ar, ai);
  float pr = ar, pi = ai;                      // a^64 via 6 complex squarings
#pragma unroll
  for (int i = 0; i < 6; ++i) { float nr = pr*pr - pi*pi; float ni = 2.f*pr*pi; pr = nr; pi = ni; }
  float hr = 0.f, hi = 0.f;
  for (int c = 0; c < NC; ++c) {
    int j = (b * NC + c) * H_DIM + h;
    Hr[j] = hr; Hi[j] = hi;
    float sr = Sr[j], si = Si[j];
    float nr = pr * hr - pi * hi + sr;
    float ni = pr * hi + pi * hr + si;
    hr = nr; hi = ni;
  }
}

// pass 3: replay chunk with true initial state, emit y (bf16 for GEMM2)
__global__ void scan_emit(const float* __restrict__ v,
                          const float* __restrict__ Alog, const float* __restrict__ Ath,
                          const float* __restrict__ Brv,  const float* __restrict__ Biv,
                          const float* __restrict__ Crv,  const float* __restrict__ Civ,
                          const float* __restrict__ Hr,   const float* __restrict__ Hi,
                          bf16_t* __restrict__ y) {
  int idx = blockIdx.x * 256 + threadIdx.x;
  int h  = idx & (H_DIM - 1);
  int bc = idx >> 10;
  int b = bc >> 6, c = bc & (NC - 1);
  float ar, ai; s4d_coef(Alog, Ath, h, ar, ai);
  float br = Brv[h], bi = Biv[h];
  float cr = Crv[h], ci = Civ[h];
  long base = ((long)b * T_DIM + c * TC) * H_DIM + h;
  float hr = Hr[idx], hi = Hi[idx];
  for (int tl = 0; tl < TC; ++tl) {
    float vv = v[base + (long)tl * H_DIM];
    float nr = ar * hr - ai * hi + br * vv;
    float ni = ar * hi + ai * hr + bi * vv;
    hr = nr; hi = ni;
    y[base + (long)tl * H_DIM] = (__bf16)(cr * hr - ci * hi);
  }
}

// ---------------------------------------------------------------------------
extern "C" void kernel_launch(void* const* d_in, const int* in_sizes, int n_in,
                              void* d_out, int out_size, void* d_ws, size_t ws_size,
                              hipStream_t stream) {
  (void)in_sizes; (void)n_in; (void)out_size; (void)ws_size;
  const float* x    = (const float*)d_in[0];
  const float* Win  = (const float*)d_in[1];
  const float* bin  = (const float*)d_in[2];
  const float* Wout = (const float*)d_in[3];
  const float* bout = (const float*)d_in[4];
  const float* Alog = (const float*)d_in[5];
  const float* Ath  = (const float*)d_in[6];
  const float* Br   = (const float*)d_in[7];
  const float* Bi   = (const float*)d_in[8];
  const float* Cr   = (const float*)d_in[9];
  const float* Ci   = (const float*)d_in[10];
  const float* Dv   = (const float*)d_in[11];

  char* ws = (char*)d_ws;
  size_t off = 0;
  auto carve = [&](size_t bytes) -> void* {
    void* p = ws + off;
    off += (bytes + 255) & ~(size_t)255;
    return p;
  };
  bf16_t* u_bf    = (bf16_t*)carve((size_t)M_DIM * CF * 2);       // 16.8 MB
  bf16_t* win_bf  = (bf16_t*)carve((size_t)H_DIM * CF * 2);       //  1.0 MB
  bf16_t* wout_bf = (bf16_t*)carve((size_t)CF * H_DIM * 2);       //  1.0 MB
  float*  vbuf    = (float*) carve((size_t)M_DIM * H_DIM * 4);    // 67.1 MB
  bf16_t* ybuf    = (bf16_t*)carve((size_t)M_DIM * H_DIM * 2);    // 33.6 MB
  float*  Srb     = (float*) carve((size_t)B_DIM * NC * H_DIM * 4);
  float*  Sib     = (float*) carve((size_t)B_DIM * NC * H_DIM * 4);
  float*  Hrb     = (float*) carve((size_t)B_DIM * NC * H_DIM * 4);
  float*  Hib     = (float*) carve((size_t)B_DIM * NC * H_DIM * 4);

  // 1. conversions
  conv_u_kernel<<<(M_DIM * CF / 4) / 256, 256, 0, stream>>>(x, u_bf);
  conv_f32_bf16<<<(H_DIM * CF / 4 + 255) / 256, 256, 0, stream>>>(Win, win_bf, H_DIM * CF / 4);
  conv_f32_bf16<<<(CF * H_DIM / 4 + 255) / 256, 256, 0, stream>>>(Wout, wout_bf, CF * H_DIM / 4);

  // 2. v = u @ Win^T + b_in   (WMMA bf16, f32 accumulate, async-LDS pipeline)
  gemm_wmma<CF, H_DIM, 0><<<dim3(M_DIM / 128, H_DIM / 128), 256, 0, stream>>>(
      u_bf, win_bf, bin, vbuf, nullptr, nullptr);

  // 3. chunk-parallel S4D scan
  scan_local<<<(B_DIM * NC * H_DIM) / 256, 256, 0, stream>>>(vbuf, Alog, Ath, Br, Bi, Srb, Sib);
  scan_combine<<<(B_DIM * H_DIM) / 256, 256, 0, stream>>>(Srb, Sib, Alog, Ath, Hrb, Hib);
  scan_emit<<<(B_DIM * NC * H_DIM) / 256, 256, 0, stream>>>(
      vbuf, Alog, Ath, Br, Bi, Cr, Ci, Hrb, Hib, ybuf);

  // 4. out = y @ Wout^T + b_out + D*u  (fused permuted store)
  gemm_wmma<H_DIM, CF, 1><<<dim3(M_DIM / 128, CF / 128), 256, 0, stream>>>(
      ybuf, wout_bf, bout, (float*)d_out, x, Dv);
}
